// CharRNN_72507637891101
// MI455X (gfx1250) — compile-verified, matrix operand order checked by reading, not run on hardware
//
#include <hip/hip_runtime.h>
#include <hip/hip_bf16.h>
#include <math.h>

typedef __attribute__((ext_vector_type(16))) __bf16 v16bf;
typedef __attribute__((ext_vector_type(8)))  __bf16 v8bf;
typedef __attribute__((ext_vector_type(8)))  float  v8f;
typedef __attribute__((ext_vector_type(4)))  int    v4i;

#define B_  64
#define T_  256
#define E_  128
#define H_  1024
#define V_  16384
#define BT_ (B_ * T_)

#ifndef __has_builtin
#define __has_builtin(x) 0
#endif
#if __has_builtin(__builtin_amdgcn_global_load_async_to_lds_b128) && \
    __has_builtin(__builtin_amdgcn_s_wait_asynccnt)
#define USE_ASYNC_LDS 1
#else
#define USE_ASYNC_LDS 0
#endif

#define AS1 __attribute__((address_space(1)))
#define AS3 __attribute__((address_space(3)))

#define WMMA_BF16(a, b, c) \
    __builtin_amdgcn_wmma_f32_16x16x32_bf16(false, (a), false, (b), (short)0, (c), false, false)

// Load one 16-lane-row fragment half-pair (16 bf16) from a row-major row.
// Per ISA layout: elems 0..7 = K (hlf*8 .. +7), elems 8..15 = K (16+hlf*8 .. +7).
__device__ __forceinline__ v16bf load_frag_row(const __bf16* row, int hlf) {
    v8bf lo = *(const v8bf*)(row + hlf * 8);
    v8bf hi = *(const v8bf*)(row + 16 + hlf * 8);
    return __builtin_shufflevector(lo, hi, 0, 1, 2, 3, 4, 5, 6, 7,
                                   8, 9, 10, 11, 12, 13, 14, 15);
}

// ---------------------------------------------------------------------------
// f32 [K,N] -> bf16 transposed [N,K]  (one-time; makes all WMMA B-fragment
// loads contiguous 16B per lane)
// ---------------------------------------------------------------------------
__global__ void cvt_t_kernel(const float* __restrict__ s, __bf16* __restrict__ d,
                             int K, int N) {
    size_t total = (size_t)K * N;
    size_t i = (size_t)blockIdx.x * blockDim.x + threadIdx.x;
    size_t stride = (size_t)gridDim.x * blockDim.x;
    for (; i < total; i += stride) {
        int k = (int)(i / N), n = (int)(i % N);
        d[(size_t)n * K + k] = (__bf16)s[i];
    }
}

// ---------------------------------------------------------------------------
// Embedding gather: xseq[t, b, :] = bf16(embedding[input[b, t], :])
// ---------------------------------------------------------------------------
__global__ __launch_bounds__(E_) void embed_kernel(const int* __restrict__ ids,
                                                   const float* __restrict__ emb,
                                                   __bf16* __restrict__ xseq) {
    int bt = blockIdx.x;          // flat b*T + t
    int b = bt >> 8, t = bt & 255;
    int e = threadIdx.x;
    int id = ids[bt];
    xseq[((size_t)t * B_ + b) * E_ + e] = (__bf16)emb[(size_t)id * E_ + e];
}

// ---------------------------------------------------------------------------
// GRU gate GEMM + sigmoid epilogue.
//   C[64, 2048] = concat(X, Hbf) @ Wg + bg ; r-half -> rh = sigm*h (bf16),
//                                            u-half -> u (f32)
// WgT is transposed [2048, K]. Split K-loop (x-part / h-part, both multiples
// of 64) + dual accumulators to break WMMA->WMMA hazard chains.
// grid (4, 32), block 128 (4 waves, one 16x16 tile each).
// ---------------------------------------------------------------------------
__global__ __launch_bounds__(128) void gru_gates_kernel(
    const __bf16* __restrict__ X, int ldx, int Esz,
    const __bf16* __restrict__ Hbf, const float* __restrict__ Hf,
    const __bf16* __restrict__ WgT, const float* __restrict__ bg,
    __bf16* __restrict__ rh, float* __restrict__ u, int K)
{
    const int lane = threadIdx.x & 31, wave = threadIdx.x >> 5;
    const int hlf = lane >> 4, nl = lane & 15;
    const int m0 = blockIdx.x * 16;
    const int n0 = (blockIdx.y * 4 + wave) * 16;
    const __bf16* bwrow = WgT + (size_t)(n0 + nl) * K;
    const __bf16* xrow  = X   + (size_t)(m0 + nl) * ldx;
    const __bf16* hrow  = Hbf + (size_t)(m0 + nl) * H_;

    v8f acc0 = {}, acc1 = {};
    for (int k0 = 0; k0 < Esz; k0 += 64) {
        v16bf a0 = load_frag_row(xrow + k0, hlf);
        v16bf b0 = load_frag_row(bwrow + k0, hlf);
        v16bf a1 = load_frag_row(xrow + k0 + 32, hlf);
        v16bf b1 = load_frag_row(bwrow + k0 + 32, hlf);
        acc0 = WMMA_BF16(a0, b0, acc0);
        acc1 = WMMA_BF16(a1, b1, acc1);
    }
    for (int k0 = Esz; k0 < K; k0 += 64) {
        v16bf a0 = load_frag_row(hrow + (k0 - Esz), hlf);
        v16bf b0 = load_frag_row(bwrow + k0, hlf);
        v16bf a1 = load_frag_row(hrow + (k0 - Esz) + 32, hlf);
        v16bf b1 = load_frag_row(bwrow + k0 + 32, hlf);
        acc0 = WMMA_BF16(a0, b0, acc0);
        acc1 = WMMA_BF16(a1, b1, acc1);
    }
    v8f acc = acc0 + acc1;
#pragma unroll
    for (int v = 0; v < 8; ++v) {
        int m = m0 + v + hlf * 8;
        int n = n0 + nl;
        float s = 1.0f / (1.0f + __expf(-(acc[v] + bg[n])));
        if (n < H_) rh[m * H_ + n] = (__bf16)(s * Hf[m * H_ + n]);   // r * h
        else        u[m * H_ + (n - H_)] = s;                        // update gate
    }
}

// ---------------------------------------------------------------------------
// GRU candidate GEMM + state-update epilogue.
//   c = tanh(concat(X, rh) @ Wc + bc);  h = u*h + (1-u)*c
// WcT is transposed [1024, K]; grid (4, 16), block 128.
// ---------------------------------------------------------------------------
__global__ __launch_bounds__(128) void gru_cand_kernel(
    const __bf16* __restrict__ X, int ldx, int Esz,
    const __bf16* __restrict__ RH, const float* __restrict__ u,
    float* __restrict__ Hf, __bf16* __restrict__ Hbf,
    const __bf16* __restrict__ WcT, const float* __restrict__ bc, int K,
    __bf16* __restrict__ ys, int t)
{
    const int lane = threadIdx.x & 31, wave = threadIdx.x >> 5;
    const int hlf = lane >> 4, nl = lane & 15;
    const int m0 = blockIdx.x * 16;
    const int n0 = (blockIdx.y * 4 + wave) * 16;
    const __bf16* bwrow = WcT + (size_t)(n0 + nl) * K;
    const __bf16* xrow  = X  + (size_t)(m0 + nl) * ldx;
    const __bf16* rrow  = RH + (size_t)(m0 + nl) * H_;

    v8f acc0 = {}, acc1 = {};
    for (int k0 = 0; k0 < Esz; k0 += 64) {
        v16bf a0 = load_frag_row(xrow + k0, hlf);
        v16bf b0 = load_frag_row(bwrow + k0, hlf);
        v16bf a1 = load_frag_row(xrow + k0 + 32, hlf);
        v16bf b1 = load_frag_row(bwrow + k0 + 32, hlf);
        acc0 = WMMA_BF16(a0, b0, acc0);
        acc1 = WMMA_BF16(a1, b1, acc1);
    }
    for (int k0 = Esz; k0 < K; k0 += 64) {
        v16bf a0 = load_frag_row(rrow + (k0 - Esz), hlf);
        v16bf b0 = load_frag_row(bwrow + k0, hlf);
        v16bf a1 = load_frag_row(rrow + (k0 - Esz) + 32, hlf);
        v16bf b1 = load_frag_row(bwrow + k0 + 32, hlf);
        acc0 = WMMA_BF16(a0, b0, acc0);
        acc1 = WMMA_BF16(a1, b1, acc1);
    }
    v8f acc = acc0 + acc1;
#pragma unroll
    for (int v = 0; v < 8; ++v) {
        int m = m0 + v + hlf * 8;
        int n = n0 + nl;
        float c = tanhf(acc[v] + bc[n]);
        float uu = u[m * H_ + n];
        float hn = uu * Hf[m * H_ + n] + (1.0f - uu) * c;
        Hf[m * H_ + n]  = hn;
        Hbf[m * H_ + n] = (__bf16)hn;
        if (ys) ys[((size_t)m * T_ + t) * H_ + n] = (__bf16)hn;  // batch-major flatten
    }
}

// ---------------------------------------------------------------------------
// Logits GEMM: out[BT, V] = ys[BT, H] @ Wsm[H, V] + bias, with WsmT [V, H].
// Block 256 threads (8 waves): 128(M) x 64(N) tile.
// B tile staged through LDS in fragment-friendly [64 n][32 k] layout using
// the gfx1250 async global->LDS DMA path (ASYNCcnt) when available.
// 4 independent accumulators per wave -> no WMMA->WMMA hazard NOPs.
// ---------------------------------------------------------------------------
__global__ __launch_bounds__(256) void logits_kernel(
    const __bf16* __restrict__ ysb, const __bf16* __restrict__ WsmT,
    const float* __restrict__ bias, float* __restrict__ out)
{
    __shared__ __bf16 Bs[64 * 32];
    const int tid = threadIdx.x;
    const int wave = tid >> 5, lane = tid & 31;
    const int hlf = lane >> 4, nl = lane & 15;
    const int m0 = blockIdx.x * 128 + wave * 16;
    const int n0 = blockIdx.y * 64;
    const int srow = tid >> 2;           // 0..63  (n within block tile)
    const int scol = (tid & 3) * 8;      // 0,8,16,24 (k within tile)

    v8f acc[4] = {};
    for (int k0 = 0; k0 < H_; k0 += 32) {
        __syncthreads();   // previous tile's reads complete
        {
            const __bf16* gsrc = WsmT + (size_t)(n0 + srow) * H_ + k0 + scol;
            __bf16* ldst = &Bs[srow * 32 + scol];
#if USE_ASYNC_LDS
            __builtin_amdgcn_global_load_async_to_lds_b128(
                (AS1 v4i*)(void*)gsrc, (AS3 v4i*)(void*)ldst, 0, 0);
            __builtin_amdgcn_s_wait_asynccnt(0);
#else
            *(v8bf*)ldst = *(const v8bf*)gsrc;
#endif
            if (k0 + 32 < H_)
                __builtin_prefetch(gsrc + 32, 0, 1);
        }
        __syncthreads();   // staged tile visible

        v16bf a = load_frag_row(ysb + (size_t)(m0 + nl) * H_ + k0, hlf);
#pragma unroll
        for (int nt = 0; nt < 4; ++nt) {
            v16bf b = load_frag_row(&Bs[(nt * 16 + nl) * 32], hlf);
            acc[nt] = WMMA_BF16(a, b, acc[nt]);
        }
    }
#pragma unroll
    for (int nt = 0; nt < 4; ++nt)
#pragma unroll
        for (int v = 0; v < 8; ++v) {
            int m = m0 + v + hlf * 8;
            int n = n0 + nt * 16 + nl;
            out[(size_t)m * V_ + n] = acc[nt][v] + bias[n];
        }
}

// ---------------------------------------------------------------------------
// Fused one-pass log-softmax + NLL per row (reads 1 GiB logits exactly once).
// ---------------------------------------------------------------------------
__global__ __launch_bounds__(256) void row_nll_kernel(
    const float* __restrict__ logits, const int* __restrict__ tgt,
    float* __restrict__ nll)
{
    __shared__ float smax[256], ssum[256];
    const int row = blockIdx.x, tid = threadIdx.x;
    const float* rp = logits + (size_t)row * V_;
    float mx = -3.4e38f, sm = 0.0f;
    for (int j = tid; j < V_; j += 256) {
        float x = rp[j];
        float nm = fmaxf(mx, x);
        sm = sm * __expf(mx - nm) + __expf(x - nm);
        mx = nm;
    }
    smax[tid] = mx; ssum[tid] = sm;
    __syncthreads();
    for (int s = 128; s > 0; s >>= 1) {
        if (tid < s) {
            float m2 = smax[tid + s], s2 = ssum[tid + s];
            float nm = fmaxf(smax[tid], m2);
            ssum[tid] = ssum[tid] * __expf(smax[tid] - nm) + s2 * __expf(m2 - nm);
            smax[tid] = nm;
        }
        __syncthreads();
    }
    if (tid == 0) {
        float lse = smax[0] + __logf(ssum[0]);
        nll[row] = lse - rp[tgt[row]];
    }
}

__global__ __launch_bounds__(256) void loss_mean_kernel(const float* __restrict__ nll,
                                                        float* __restrict__ out) {
    __shared__ float s[256];
    int tid = threadIdx.x;
    float acc = 0.0f;
    for (int i = tid; i < BT_; i += 256) acc += nll[i];
    s[tid] = acc; __syncthreads();
    for (int k = 128; k > 0; k >>= 1) { if (tid < k) s[tid] += s[tid + k]; __syncthreads(); }
    if (tid == 0) out[0] = s[0] / (float)BT_;
}

// ---------------------------------------------------------------------------
extern "C" void kernel_launch(void* const* d_in, const int* in_sizes, int n_in,
                              void* d_out, int out_size, void* d_ws, size_t ws_size,
                              hipStream_t stream) {
    const int*   ids = (const int*)d_in[0];
    const int*   tgt = (const int*)d_in[1];
    const float* emb = (const float*)d_in[2];
    const float* Wg1 = (const float*)d_in[3];
    const float* bg1 = (const float*)d_in[4];
    const float* Wc1 = (const float*)d_in[5];
    const float* bc1 = (const float*)d_in[6];
    const float* Wg2 = (const float*)d_in[7];
    const float* bg2 = (const float*)d_in[8];
    const float* Wc2 = (const float*)d_in[9];
    const float* bc2 = (const float*)d_in[10];
    const float* Wsm = (const float*)d_in[11];
    const float* bsm = (const float*)d_in[12];

    float* outLogits = (float*)d_out;
    float* outLoss   = outLogits + (size_t)BT_ * V_;

    // workspace carve-out (256B aligned)
    char* ws = (char*)d_ws;
    size_t off = 0;
    auto carve = [&](size_t bytes) -> void* {
        void* p = ws + off;
        off += (bytes + 255) & ~(size_t)255;
        return p;
    };
    const int EH1 = E_ + H_;           // 1152
    const int EH2 = 2 * H_;            // 2048
    __bf16* Wg1T = (__bf16*)carve((size_t)EH1 * 2 * H_ * 2);   // [2H, EH1]
    __bf16* Wc1T = (__bf16*)carve((size_t)EH1 * H_ * 2);       // [H, EH1]
    __bf16* Wg2T = (__bf16*)carve((size_t)EH2 * 2 * H_ * 2);   // [2H, EH2]
    __bf16* Wc2T = (__bf16*)carve((size_t)EH2 * H_ * 2);       // [H, EH2]
    __bf16* WsmT = (__bf16*)carve((size_t)H_ * V_ * 2);        // [V, H]
    __bf16* xseq = (__bf16*)carve((size_t)T_ * B_ * E_ * 2);
    __bf16* ysb  = (__bf16*)carve((size_t)BT_ * H_ * 2);
    float*  h1f  = (float*)carve((size_t)B_ * H_ * 4);
    float*  h2f  = (float*)carve((size_t)B_ * H_ * 4);
    __bf16* h1b  = (__bf16*)carve((size_t)B_ * H_ * 2);
    __bf16* h2b  = (__bf16*)carve((size_t)B_ * H_ * 2);
    __bf16* rh1  = (__bf16*)carve((size_t)B_ * H_ * 2);
    __bf16* rh2  = (__bf16*)carve((size_t)B_ * H_ * 2);
    float*  u1   = (float*)carve((size_t)B_ * H_ * 4);
    float*  u2   = (float*)carve((size_t)B_ * H_ * 4);
    float*  nll  = (float*)carve((size_t)BT_ * 4);

    // 1) weights -> bf16, transposed (then L2-resident for the whole scan)
    auto cvtT = [&](const float* s, __bf16* d, int K, int N) {
        int n = K * N;
        cvt_t_kernel<<<(n + 255) / 256, 256, 0, stream>>>(s, d, K, N);
    };
    cvtT(Wg1, Wg1T, EH1, 2 * H_);
    cvtT(Wc1, Wc1T, EH1, H_);
    cvtT(Wg2, Wg2T, EH2, 2 * H_);
    cvtT(Wc2, Wc2T, EH2, H_);
    cvtT(Wsm, WsmT, H_, V_);

    // 2) embedding gather (time-major)
    embed_kernel<<<BT_, E_, 0, stream>>>(ids, emb, xseq);

    // 3) zero hidden state
    (void)hipMemsetAsync(h1f, 0, (size_t)B_ * H_ * 4, stream);
    (void)hipMemsetAsync(h2f, 0, (size_t)B_ * H_ * 4, stream);
    (void)hipMemsetAsync(h1b, 0, (size_t)B_ * H_ * 2, stream);
    (void)hipMemsetAsync(h2b, 0, (size_t)B_ * H_ * 2, stream);

    // 4) sequential 2-layer GRU scan
    for (int t = 0; t < T_; ++t) {
        const __bf16* xt = xseq + (size_t)t * B_ * E_;
        gru_gates_kernel<<<dim3(4, 32), 128, 0, stream>>>(
            xt, E_, E_, h1b, h1f, Wg1T, bg1, rh1, u1, EH1);
        gru_cand_kernel<<<dim3(4, 16), 128, 0, stream>>>(
            xt, E_, E_, rh1, u1, h1f, h1b, Wc1T, bc1, EH1, (__bf16*)nullptr, t);
        gru_gates_kernel<<<dim3(4, 32), 128, 0, stream>>>(
            h1b, H_, H_, h2b, h2f, Wg2T, bg2, rh2, u2, EH2);
        gru_cand_kernel<<<dim3(4, 16), 128, 0, stream>>>(
            h1b, H_, H_, rh2, u2, h2f, h2b, Wc2T, bc2, EH2, ysb, t);
    }

    // 5) logits GEMM (550 GFLOP; HBM floor = 1 GiB store)
    logits_kernel<<<dim3(BT_ / 128, V_ / 64), 256, 0, stream>>>(ysb, WsmT, bsm, outLogits);

    // 6) fused log-softmax + NLL, then mean -> loss
    row_nll_kernel<<<BT_, 256, 0, stream>>>(outLogits, tgt, nll);
    loss_mean_kernel<<<1, 256, 0, stream>>>(nll, outLoss);
}
